// GraphTransformer_12833362280632
// MI455X (gfx1250) — compile-verified, matrix operand order checked by reading, not run on hardware
//
#include <hip/hip_runtime.h>
#include <stdint.h>

// ---------------------------------------------------------------------------
// Graph Transformer forward for MI455X (gfx1250, wave32).
// bf16 activations + pre-transposed bf16 weights; GEMM inner loop is pure
// global_load_b128 + v_wmma_f32_16x16x32_bf16 with compile-time K/N (no
// guards, no conversions, immediate-offset addressing).
// ---------------------------------------------------------------------------

#define NN   384
#define NSQ  (NN * NN)

typedef __bf16   bf16x16 __attribute__((ext_vector_type(16)));
typedef float    f32x8   __attribute__((ext_vector_type(8)));

__device__ __forceinline__ uint16_t f2bf(float f) {
    uint32_t u = __builtin_bit_cast(uint32_t, f);
    u += 0x7FFFu + ((u >> 16) & 1u);          // round-to-nearest-even
    return (uint16_t)(u >> 16);
}
__device__ __forceinline__ float bf2f(uint16_t h) {
    return __builtin_bit_cast(float, (uint32_t)h << 16);
}

// 16x32 bf16 fragment (ISA 7.12.2): for lane (lm, hi), VGPRs 0..3 hold
// K = hi*8 .. hi*8+7 and VGPRs 4..7 hold K = 16+hi*8 .. 16+hi*8+7, two
// bf16 per VGPR -> exactly two aligned 16-byte loads at p and p+16 elems.
__device__ __forceinline__ bf16x16 load_frag(const uint16_t* __restrict__ p) {
    union { uint4 q[2]; bf16x16 v; } u;
    u.q[0] = *reinterpret_cast<const uint4*>(p);
    u.q[1] = *reinterpret_cast<const uint4*>(p + 16);
    return u.v;
}

template <int N, bool RELU>
__device__ __forceinline__ void
store_tile(const f32x8& acc, uint16_t* __restrict__ C, const float* __restrict__ bias,
           int r0, int c0, int lm, int hi) {
    const int c = c0 + lm;                 // C/D layout: N = lane%16
    const float bv = bias[c];
#pragma unroll
    for (int v = 0; v < 8; ++v) {          //             M = vgpr + 8*(lane>=16)
        float val = acc[v] + bv;
        if (RELU) val = fmaxf(val, 0.f);
        C[(size_t)(r0 + hi * 8 + v) * N + c] = f2bf(val);
    }
}

// C[M,N] = act(A[M,K] @ Bt[N,K]^T + bias), all bf16, fp32 accum.
// 128 threads = 4 waves; block tile 64x64; wave tile 32x32 (2x2 WMMA tiles).
// Requires: M % 64 == 0, N % 64 == 0, K % 32 == 0 (stems are zero-padded).
template <int K, int N, bool RELU>
__global__ __launch_bounds__(128) void
gemm_k(const uint16_t* __restrict__ A, const uint16_t* __restrict__ Bt,
       const float* __restrict__ bias, uint16_t* __restrict__ C, int M) {
    const int lane = threadIdx.x & 31;
    const int wave = threadIdx.x >> 5;
    const int lm   = lane & 15;
    const int hi   = lane >> 4;
    const int m0 = blockIdx.y * 64 + (wave >> 1) * 32;
    const int n0 = blockIdx.x * 64 + (wave & 1) * 32;

    const uint16_t* pa0 = A  + (size_t)(m0 + lm)      * K + hi * 8;
    const uint16_t* pa1 = A  + (size_t)(m0 + 16 + lm) * K + hi * 8;
    const uint16_t* pb0 = Bt + (size_t)(n0 + lm)      * K + hi * 8;
    const uint16_t* pb1 = Bt + (size_t)(n0 + 16 + lm) * K + hi * 8;

    f32x8 acc00 = {}, acc01 = {}, acc10 = {}, acc11 = {};

#pragma unroll
    for (int k0 = 0; k0 < K; k0 += 32) {
        bf16x16 a0 = load_frag(pa0 + k0);
        bf16x16 a1 = load_frag(pa1 + k0);
        bf16x16 b0 = load_frag(pb0 + k0);
        bf16x16 b1 = load_frag(pb1 + k0);
        acc00 = __builtin_amdgcn_wmma_f32_16x16x32_bf16(false, a0, false, b0, (short)0, acc00, false, false);
        acc01 = __builtin_amdgcn_wmma_f32_16x16x32_bf16(false, a0, false, b1, (short)0, acc01, false, false);
        acc10 = __builtin_amdgcn_wmma_f32_16x16x32_bf16(false, a1, false, b0, (short)0, acc10, false, false);
        acc11 = __builtin_amdgcn_wmma_f32_16x16x32_bf16(false, a1, false, b1, (short)0, acc11, false, false);
    }
    store_tile<N, RELU>(acc00, C, bias, m0,      n0,      lm, hi);
    store_tile<N, RELU>(acc01, C, bias, m0,      n0 + 16, lm, hi);
    store_tile<N, RELU>(acc10, C, bias, m0 + 16, n0,      lm, hi);
    store_tile<N, RELU>(acc11, C, bias, m0 + 16, n0 + 16, lm, hi);
}

template <int K, int N, bool RELU>
static inline void launch_gemm(const uint16_t* A, const uint16_t* Bt, const float* bias,
                               uint16_t* C, int M, hipStream_t s) {
    dim3 g((unsigned)(N / 64), (unsigned)(M / 64)), b(128);
    gemm_k<K, N, RELU><<<g, b, 0, s>>>(A, Bt, bias, C, M);
}

// ---------------- conversion / elementwise / reduction kernels ------------

// fp32 [K,N] weight -> bf16 transposed [N,Kpad], zero-padded in K.
__global__ void tc_k(const float* __restrict__ in, uint16_t* __restrict__ out,
                     int K, int N, int Kpad) {
    int i = blockIdx.x * blockDim.x + threadIdx.x;
    if (i >= N * Kpad) return;
    int n = i / Kpad, k = i % Kpad;
    out[i] = (k < K) ? f2bf(in[(size_t)k * N + n]) : (uint16_t)0;
}

__global__ void f2b_k(const float* __restrict__ in, uint16_t* __restrict__ out, size_t n) {
    size_t i = (size_t)blockIdx.x * blockDim.x + threadIdx.x;
    if (i < n) out[i] = f2bf(in[i]);
}

// fp32 [rows,Cin] -> bf16 [rows,Cpad] zero-padded columns (stem K padding).
__global__ void cvt_pad_k(const float* __restrict__ in, uint16_t* __restrict__ out,
                          size_t rows, int Cin, int Cpad) {
    size_t i = (size_t)blockIdx.x * blockDim.x + threadIdx.x;
    if (i >= rows * (size_t)Cpad) return;
    size_t r = i / Cpad;
    int c = (int)(i % Cpad);
    out[i] = (c < Cin) ? f2bf(in[r * Cin + c]) : (uint16_t)0;
}

__global__ void add_bf_k(uint16_t* __restrict__ a, const uint16_t* __restrict__ b, size_t n) {
    size_t i = (size_t)blockIdx.x * blockDim.x + threadIdx.x;
    if (i < n) a[i] = f2bf(bf2f(a[i]) + bf2f(b[i]));
}

__global__ void skip_k(const uint16_t* __restrict__ e, float* __restrict__ skip, int n2) {
    int i = blockIdx.x * blockDim.x + threadIdx.x;
    if (i < n2) skip[i] = bf2f(e[(size_t)i * 128]);   // e[..., :1]
}

// out = LayerNorm(a + b) * gamma + beta over rows of 128, bf16 in/out.
// 8 waves/block, one wave per row, 4 floats/lane, shfl_xor wave32 reduction.
__global__ __launch_bounds__(256) void
ln_res_k(const uint16_t* __restrict__ a, const uint16_t* __restrict__ b,
         const float* __restrict__ gamma, const float* __restrict__ beta,
         uint16_t* __restrict__ out, int rows) {
    const int lane = threadIdx.x & 31, wv = threadIdx.x >> 5;
    const int row = blockIdx.x * 8 + wv;
    if (row >= rows) return;
    const size_t base = (size_t)row * 128 + lane * 4;
    float x[4], s1 = 0.f, s2 = 0.f;
#pragma unroll
    for (int t = 0; t < 4; ++t) {
        x[t] = bf2f(a[base + t]) + bf2f(b[base + t]);
        s1 += x[t];
        s2 += x[t] * x[t];
    }
#pragma unroll
    for (int o = 16; o > 0; o >>= 1) {
        s1 += __shfl_xor(s1, o, 32);
        s2 += __shfl_xor(s2, o, 32);
    }
    const float m  = s1 * (1.f / 128.f);
    const float rs = rsqrtf(s2 * (1.f / 128.f) - m * m + 1e-5f);
#pragma unroll
    for (int t = 0; t < 4; ++t) {
        int c = lane * 4 + t;
        out[base + t] = f2bf((x[t] - m) * rs * gamma[c] + beta[c]);
    }
}

// Per-feature cross attention: out[i,hd] = sum_j softmax_j(Q[i,hd]*K[j,hd]/4)*V[j,hd]
__global__ __launch_bounds__(128) void
ca_k(const uint16_t* __restrict__ Q, const uint16_t* __restrict__ K,
     const uint16_t* __restrict__ V, uint16_t* __restrict__ out) {
    const int i = blockIdx.x, t = threadIdx.x;
    const float q = bf2f(Q[i * 128 + t]) * 0.25f;   // 1/sqrt(DF)
    float m = -3.4e38f, s = 0.f, acc = 0.f;
    for (int j = 0; j < NN; ++j) {
        float l  = q * bf2f(K[j * 128 + t]);
        float nm = fmaxf(m, l);
        float sc = __expf(m - nm);
        float w  = __expf(l - nm);
        s   = s * sc + w;
        acc = acc * sc + w * bf2f(V[j * 128 + t]);
        m   = nm;
    }
    out[i * 128 + t] = f2bf(acc / s);
}

// Y = (Q[i]*K[j]/4)*(E1+1)+E2, written in place over E1.
__global__ __launch_bounds__(128) void
y_k(const uint16_t* __restrict__ Q, const uint16_t* __restrict__ K,
    uint16_t* __restrict__ E1io, const uint16_t* __restrict__ E2) {
    const size_t ij = blockIdx.x;
    const int i = (int)(ij / NN), j = (int)(ij % NN);
    const int t = threadIdx.x;
    const size_t o = ij * 128 + t;
    float y = (bf2f(Q[i * 128 + t]) * bf2f(K[j * 128 + t]) * 0.25f)
                  * (bf2f(E1io[o]) + 1.f) + bf2f(E2[o]);
    E1io[o] = f2bf(y);
}

// wV[i,hd] = sum_j softmax_j(Y[i,j,hd]) * V[j,hd]   (single pass over Y)
__global__ __launch_bounds__(128) void
attnv_k(const uint16_t* __restrict__ Y, const uint16_t* __restrict__ V,
        uint16_t* __restrict__ wV) {
    const int i = blockIdx.x, t = threadIdx.x;
    float m = -3.4e38f, s = 0.f, acc = 0.f;
    for (int j = 0; j < NN; ++j) {
        float l  = bf2f(Y[((size_t)i * NN + j) * 128 + t]);
        float nm = fmaxf(m, l);
        float sc = __expf(m - nm);
        float w  = __expf(l - nm);
        s   = s * sc + w;
        acc = acc * sc + w * bf2f(V[j * 128 + t]);
        m   = nm;
    }
    wV[i * 128 + t] = f2bf(acc / s);
}

// out[ij] = dot(T[ij,:128], W2[:,0]) + b2 + skip[ij]   (fp32 out)
__global__ __launch_bounds__(256) void
final_k(const uint16_t* __restrict__ T, const float* __restrict__ W2,
        const float* __restrict__ b2, const float* __restrict__ skip,
        float* __restrict__ out, int rows) {
    const int lane = threadIdx.x & 31, wv = threadIdx.x >> 5;
    const int row = blockIdx.x * 8 + wv;
    if (row >= rows) return;
    const size_t base = (size_t)row * 128 + lane * 4;
    float s = 0.f;
#pragma unroll
    for (int t = 0; t < 4; ++t) s += bf2f(T[base + t]) * W2[lane * 4 + t];
#pragma unroll
    for (int o = 16; o > 0; o >>= 1) s += __shfl_xor(s, o, 32);
    if (lane == 0) out[row] = s + b2[0] + skip[row];
}

// ---------------------------------------------------------------------------

extern "C" void kernel_launch(void* const* d_in, const int* in_sizes, int n_in,
                              void* d_out, int out_size, void* d_ws, size_t ws_size,
                              hipStream_t stream) {
    (void)in_sizes; (void)n_in; (void)out_size; (void)ws_size;
    auto W = [&](int i) -> const float* { return (const float*)d_in[i]; };

    // d_in flat order (setup_inputs insertion order, params recursed in dict/tuple order):
    // 0:x 1:e 2:adj 3:n_emb 4:e_emb
    // 5..16: xin1(W,b) xin2 ein1 ein2 eout1 eout2
    // 17+30*L: gtl{L}: neb{q,k,v,e_add,e_mul,x_out,e_out}(14) linX1 linX2
    //                  normX1 normX2 linE1 linE2 normE1 normE2 (16)
    // 107+2*i: lx_i   113+2*i: le_i   119+6*i: ca_i{q,k,v}
    char* ws = (char*)d_ws;
    size_t off = 0;
    auto alloc = [&](size_t bytes) -> char* {
        char* p = ws + off;
        off += (bytes + 255) & ~(size_t)255;
        return p;
    };
    const size_t BIG = (size_t)NSQ * 128 * sizeof(uint16_t);   // 37.75 MB
    uint16_t* eA  = (uint16_t*)alloc(BIG);          // e_emb ping
    uint16_t* eB  = (uint16_t*)alloc(BIG);          // e_emb pong
    uint16_t* eh  = (uint16_t*)alloc(BIG);          // e hidden state
    uint16_t* t1  = (uint16_t*)alloc(BIG);          // E1 -> Y -> ff tmp
    uint16_t* t2  = (uint16_t*)alloc(BIG);          // E2 -> newE -> ff tmp
    float*    sk  = (float*)   alloc((size_t)NSQ * sizeof(float));
    uint16_t* x32 = (uint16_t*)alloc((size_t)NN * 32 * 2);     // x padded 16->32
    uint16_t* e32 = (uint16_t*)alloc((size_t)NSQ * 32 * 2);    // e padded 8->32
    uint16_t* nA  = (uint16_t*)alloc((size_t)NN * 128 * 2);
    uint16_t* nB  = (uint16_t*)alloc((size_t)NN * 128 * 2);
    uint16_t* xA  = (uint16_t*)alloc((size_t)NN * 128 * 2);
    uint16_t* xB  = (uint16_t*)alloc((size_t)NN * 128 * 2);
    uint16_t* Qb  = (uint16_t*)alloc((size_t)NN * 128 * 2);
    uint16_t* Kb  = (uint16_t*)alloc((size_t)NN * 128 * 2);
    uint16_t* Vb  = (uint16_t*)alloc((size_t)NN * 128 * 2);
    uint16_t* wV  = (uint16_t*)alloc((size_t)NN * 128 * 2);
    uint16_t* xff = (uint16_t*)alloc((size_t)NN * 512 * 2);
    uint16_t* xt  = (uint16_t*)alloc((size_t)NN * 128 * 2);

    // ---- pre-transpose + convert all GEMM weights to bf16 [N,Kpad] ----
    uint16_t* tw[160] = {};
    auto prep = [&](int idx, int K, int N, int Kpad) {
        uint16_t* dst = (uint16_t*)alloc((size_t)N * Kpad * 2);
        tw[idx] = dst;
        int n = N * Kpad;
        tc_k<<<(n + 255) / 256, 256, 0, stream>>>(W(idx), dst, K, N, Kpad);
    };
    prep(5, 16, 128, 32);    // xin1
    prep(7, 128, 128, 128);  // xin2
    prep(9, 8, 128, 32);     // ein1
    prep(11, 128, 128, 128); // ein2
    prep(13, 128, 128, 128); // eout1
    for (int L = 0; L < 3; ++L) {
        const int g = 17 + 30 * L, c = 119 + 6 * L;
        for (int q = 0; q < 14; q += 2) prep(g + q, 128, 128, 128); // neb (7 mats)
        prep(g + 14, 128, 512, 128);  // linX1
        prep(g + 16, 512, 128, 512);  // linX2
        prep(g + 22, 128, 128, 128);  // linE1
        prep(g + 24, 128, 128, 128);  // linE2
        prep(107 + 2 * L, 128, 128, 128); // lx
        prep(113 + 2 * L, 128, 128, 128); // le
        prep(c + 0, 128, 128, 128);   // ca q
        prep(c + 2, 128, 128, 128);   // ca k
        prep(c + 4, 128, 128, 128);   // ca v
    }

    // ---- convert fp32 inputs to bf16 activations (stems K-padded) ----
    cvt_pad_k<<<(NN * 32 + 255) / 256, 256, 0, stream>>>(W(0), x32, NN, 16, 32);
    cvt_pad_k<<<(unsigned)(((size_t)NSQ * 32 + 255) / 256), 256, 0, stream>>>(W(1), e32, NSQ, 8, 32);
    f2b_k<<<(NN * 128 + 255) / 256, 256, 0, stream>>>(W(3), nA, (size_t)NN * 128);
    f2b_k<<<(unsigned)(((size_t)NSQ * 128 + 255) / 256), 256, 0, stream>>>(W(4), eA, (size_t)NSQ * 128);

    // ---- stems ----
    launch_gemm<32, 128, true >(x32, tw[5],  W(6),  xt, NN,  stream);   // xin1+relu
    launch_gemm<128, 128, false>(xt, tw[7],  W(8),  xA, NN,  stream);   // xin2
    launch_gemm<32, 128, true >(e32, tw[9],  W(10), t1, NSQ, stream);   // ein1+relu
    launch_gemm<128, 128, false>(t1, tw[11], W(12), eh, NSQ, stream);   // ein2
    skip_k<<<(NSQ + 255) / 256, 256, 0, stream>>>(eh, sk, NSQ);

    uint16_t* xh = xA;  uint16_t* xalt = xB;
    uint16_t* nc = nA;  uint16_t* nalt = nB;
    uint16_t* ec = eA;  uint16_t* ealt = eB;
    const unsigned LNB_E = (NSQ + 7) / 8, LNB_X = (NN + 7) / 8;

    for (int L = 0; L < 3; ++L) {
        const int g = 17 + 30 * L;
        const int c = 119 + 6 * L;
        // embeddings through per-layer linears + relu
        launch_gemm<128, 128, true>(nc, tw[107 + 2 * L], W(108 + 2 * L), nalt, NN, stream);
        { uint16_t* t = nc; nc = nalt; nalt = t; }
        launch_gemm<128, 128, true>(ec, tw[113 + 2 * L], W(114 + 2 * L), ealt, NSQ, stream);
        { uint16_t* t = ec; ec = ealt; ealt = t; }
        // cross attention: x_h <- CA(x_h, n_emb)
        launch_gemm<128, 128, false>(xh, tw[c + 0], W(c + 1), Qb, NN, stream);
        launch_gemm<128, 128, false>(nc, tw[c + 2], W(c + 3), Kb, NN, stream);
        launch_gemm<128, 128, false>(nc, tw[c + 4], W(c + 5), Vb, NN, stream);
        ca_k<<<NN, 128, 0, stream>>>(Qb, Kb, Vb, xalt);
        { uint16_t* t = xh; xh = xalt; xalt = t; }
        // e_h += e_emb
        add_bf_k<<<(unsigned)(((size_t)NSQ * 128 + 255) / 256), 256, 0, stream>>>(
            eh, ec, (size_t)NSQ * 128);
        // node_edge_block
        launch_gemm<128, 128, false>(xh, tw[g + 0], W(g + 1), Qb, NN, stream);   // q
        launch_gemm<128, 128, false>(xh, tw[g + 2], W(g + 3), Kb, NN, stream);   // k
        launch_gemm<128, 128, false>(xh, tw[g + 4], W(g + 5), Vb, NN, stream);   // v
        launch_gemm<128, 128, false>(eh, tw[g + 8], W(g + 9), t1, NSQ, stream);  // E1 = e_mul
        launch_gemm<128, 128, false>(eh, tw[g + 6], W(g + 7), t2, NSQ, stream);  // E2 = e_add
        y_k<<<NSQ, 128, 0, stream>>>(Qb, Kb, t1, t2);                            // t1 <- Y
        launch_gemm<128, 128, false>(t1, tw[g + 12], W(g + 13), t2, NSQ, stream); // newE
        attnv_k<<<NN, 128, 0, stream>>>(t1, Vb, wV);
        launch_gemm<128, 128, false>(wV, tw[g + 10], W(g + 11), xalt, NN, stream); // newX
        // residual layernorms
        ln_res_k<<<LNB_X, 256, 0, stream>>>(xh, xalt, W(g + 18), W(g + 19), xh, NN);
        ln_res_k<<<LNB_E, 256, 0, stream>>>(eh, t2,   W(g + 26), W(g + 27), eh, NSQ);
        // x feed-forward
        launch_gemm<128, 512, true >(xh,  tw[g + 14], W(g + 15), xff,  NN, stream);
        launch_gemm<512, 128, false>(xff, tw[g + 16], W(g + 17), xalt, NN, stream);
        ln_res_k<<<LNB_X, 256, 0, stream>>>(xh, xalt, W(g + 20), W(g + 21), xh, NN);
        // e feed-forward
        launch_gemm<128, 128, true >(eh, tw[g + 22], W(g + 23), t1, NSQ, stream);
        launch_gemm<128, 128, false>(t1, tw[g + 24], W(g + 25), t2, NSQ, stream);
        ln_res_k<<<LNB_E, 256, 0, stream>>>(eh, t2, W(g + 28), W(g + 29), eh, NSQ);
    }

    // ---- head: e_out = lin(eout2, relu(lin(eout1, e_h))) + e_skip ----
    launch_gemm<128, 128, true>(eh, tw[13], W(14), t1, NSQ, stream);
    final_k<<<(NSQ + 7) / 8, 256, 0, stream>>>(t1, W(15), W(16), sk, (float*)d_out, NSQ);
}
